// AFT_FULL_71966472011797
// MI455X (gfx1250) — matrix operand order repaired and checked by reading, not compiled, not run on hardware
//
#include <hip/hip_runtime.h>
#include <cmath>

// ---------------------------------------------------------------------------
// AFT-Full fused kernel for MI455X (gfx1250, wave32, WMMA f16->f32).
//
// Shapes: BS=4, C=128, IC=64, T=H*W=4096.
//   q = Wq@x ; k = Wk@x ; v = Wv@x          (K=128 GEMMs)
//   num = eB@(exp(k)*v) ; den = eB@exp(k)   (K=64 GEMMs, eB=exp(B) prefolded)
//   out = Wm@(q*num/den)                    (K=64 GEMM)
// All GEMMs via v_wmma_f32_16x16x32_f16, f32 accumulate.
// B fragments hoisted: each stage's shared-K fragments loaded exactly once.
// ---------------------------------------------------------------------------

typedef _Float16 v16h __attribute__((ext_vector_type(16)));
typedef _Float16 h8   __attribute__((ext_vector_type(8)));
typedef _Float16 h2   __attribute__((ext_vector_type(2)));
typedef float    v8f  __attribute__((ext_vector_type(8)));

#define BSZ    4
#define CCH    128
#define ICH    64
#define TT     4096
#define T_TILE 64
#define NWAVE  4
#define XSTR   130   // LDS stride for x tile [t][c] (pad 2 halves)
#define SSTR   72    // LDS stride for staging [t][i] (pad 8 halves)

// f16 weight workspace layout (halves):
#define WQ_OFF 0
#define WK_OFF 8192
#define WV_OFF 16384
#define WM_OFF 24576
#define EB_OFF 32768
#define WS_TOT 36864

static __device__ inline v16h ld_frag16(const _Float16* p, int off0, int off1) {
    v16h r;
    *reinterpret_cast<h8*>(&r)       = *reinterpret_cast<const h8*>(p + off0);
    *(reinterpret_cast<h8*>(&r) + 1) = *reinterpret_cast<const h8*>(p + off1);
    return r;
}

static __device__ inline v8f wmma16(v16h a, v16h b, v8f c) {
    // (neg_a, A, neg_b, B, c_mod, C, reuse_a, reuse_b)
    return __builtin_amdgcn_wmma_f32_16x16x32_f16(false, a, false, b, (short)0, c,
                                                  false, false);
}

// --- prep: convert weights to f16, fold exp(B) -> eB ------------------------
__global__ void aft_prep(const float* __restrict__ Wq, const float* __restrict__ Wk,
                         const float* __restrict__ Wv, const float* __restrict__ Bb,
                         const float* __restrict__ Wm, _Float16* __restrict__ wsh) {
    int i = blockIdx.x * blockDim.x + threadIdx.x;
    if (i >= WS_TOT) return;
    float v;
    if      (i < WK_OFF) v = Wq[i];
    else if (i < WV_OFF) v = Wk[i - WK_OFF];
    else if (i < WM_OFF) v = Wv[i - WV_OFF];
    else if (i < EB_OFF) v = Wm[i - WM_OFF];
    else                 v = __expf(Bb[i - EB_OFF]);
    wsh[i] = (_Float16)v;
}

// --- fused main -------------------------------------------------------------
__global__ __launch_bounds__(128) void
aft_main(const float* __restrict__ xg, const _Float16* __restrict__ wsh,
         float* __restrict__ outg) {
    __shared__ _Float16 xs[T_TILE * XSTR];            // x tile, [t][c], f16
    __shared__ _Float16 stg[NWAVE * 2 * 16 * SSTR];   // per-wave ek / ekv / mid

    const int tid  = threadIdx.x;
    const int w    = tid >> 5;      // wave id: owns columns [w*16, w*16+16)
    const int lane = tid & 31;
    const int tl   = lane & 15;     // N / column within 16-wide tile
    const int hi8  = lane >> 4;     // upper-half-lane selector
    const int t0   = blockIdx.x * T_TILE;
    const int b    = blockIdx.y;

    // ---- phase 0: load x tile (b128 per lane), f32->f16, transpose to [t][c]
    {
        const int rsub = lane >> 4;          // 2 rows per wave per iteration
        const int tcol = (lane & 15) * 4;    // 4 t-columns per lane
#pragma unroll 4
        for (int c0 = w * 2; c0 < CCH; c0 += NWAVE * 2) {
            const int c = c0 + rsub;
            const float4 v4 = *reinterpret_cast<const float4*>(
                xg + (size_t)(b * CCH + c) * TT + t0 + tcol);
            xs[(tcol + 0) * XSTR + c] = (_Float16)v4.x;
            xs[(tcol + 1) * XSTR + c] = (_Float16)v4.y;
            xs[(tcol + 2) * XSTR + c] = (_Float16)v4.z;
            xs[(tcol + 3) * XSTR + c] = (_Float16)v4.w;
        }
    }
    __syncthreads();

    _Float16* ekp = stg + w * (2 * 16 * SSTR);  // exp(k) staging  [t][i]
    _Float16* evp = ekp + 16 * SSTR;            // exp(k)*v staging

    const int bxbase = (w * 16 + tl) * XSTR + 8 * hi8;   // B frag base in xs
    const int sbase  = tl * SSTR;                        // staging row base

    // ---- stage 1 B fragments (X tile): shared by q/k/v and all j-tiles ----
    v16h bx[4];
#pragma unroll
    for (int kb = 0; kb < 4; ++kb)
        bx[kb] = ld_frag16(xs, bxbase + 32 * kb, bxbase + 32 * kb + 16);

    // ---- stage 1: q = Wq@x (keep in regs) ----
    v8f qa[4];
#pragma unroll
    for (int jt = 0; jt < 4; ++jt) {
        v8f acc = {0.f, 0.f, 0.f, 0.f, 0.f, 0.f, 0.f, 0.f};
#pragma unroll
        for (int kb = 0; kb < 4; ++kb) {
            const int ao = (16 * jt + tl) * CCH + 32 * kb + 8 * hi8;
            v16h a = ld_frag16(wsh + WQ_OFF, ao, ao + 16);
            acc = wmma16(a, bx[kb], acc);
        }
        qa[jt] = acc;
    }

    // ---- stage 1: k = Wk@x ; ek = exp(k) -> regs + staging ----
    float ekr[4][8];
#pragma unroll
    for (int jt = 0; jt < 4; ++jt) {
        v8f acc = {0.f, 0.f, 0.f, 0.f, 0.f, 0.f, 0.f, 0.f};
#pragma unroll
        for (int kb = 0; kb < 4; ++kb) {
            const int ao = (16 * jt + tl) * CCH + 32 * kb + 8 * hi8;
            v16h a = ld_frag16(wsh + WK_OFF, ao, ao + 16);
            acc = wmma16(a, bx[kb], acc);
        }
#pragma unroll
        for (int r = 0; r < 8; r += 2) {
            const float e0 = __expf(acc[r]);
            const float e1 = __expf(acc[r + 1]);
            ekr[jt][r] = e0;  ekr[jt][r + 1] = e1;
            h2 pk;  pk.x = (_Float16)e0;  pk.y = (_Float16)e1;
            *reinterpret_cast<h2*>(ekp + sbase + 16 * jt + 8 * hi8 + r) = pk;
        }
    }

    // ---- stage 1: v = Wv@x ; ekv = ek*v -> staging ----
#pragma unroll
    for (int jt = 0; jt < 4; ++jt) {
        v8f acc = {0.f, 0.f, 0.f, 0.f, 0.f, 0.f, 0.f, 0.f};
#pragma unroll
        for (int kb = 0; kb < 4; ++kb) {
            const int ao = (16 * jt + tl) * CCH + 32 * kb + 8 * hi8;
            v16h a = ld_frag16(wsh + WV_OFF, ao, ao + 16);
            acc = wmma16(a, bx[kb], acc);
        }
#pragma unroll
        for (int r = 0; r < 8; r += 2) {
            h2 pk;
            pk.x = (_Float16)(ekr[jt][r]     * acc[r]);
            pk.y = (_Float16)(ekr[jt][r + 1] * acc[r + 1]);
            *reinterpret_cast<h2*>(evp + sbase + 16 * jt + 8 * hi8 + r) = pk;
        }
    }

    // ---- stage 2 B fragments (ek, ekv): shared by all j-tiles ----
    const int bo0 = sbase + 8 * hi8;
    v16h bn0 = ld_frag16(evp, bo0,      bo0 + 16);
    v16h bn1 = ld_frag16(evp, bo0 + 32, bo0 + 48);
    v16h bd0 = ld_frag16(ekp, bo0,      bo0 + 16);
    v16h bd1 = ld_frag16(ekp, bo0 + 32, bo0 + 48);

    // ---- stage 2: num = eB@ekv ; den = eB@ek ; mid = q*num/den (in regs) ----
#pragma unroll
    for (int jt = 0; jt < 4; ++jt) {
        v8f na = {0.f, 0.f, 0.f, 0.f, 0.f, 0.f, 0.f, 0.f};
        v8f da = {0.f, 0.f, 0.f, 0.f, 0.f, 0.f, 0.f, 0.f};
        const int ao0 = (16 * jt + tl) * ICH + 8 * hi8;
        v16h a0 = ld_frag16(wsh + EB_OFF, ao0,      ao0 + 16);
        v16h a1 = ld_frag16(wsh + EB_OFF, ao0 + 32, ao0 + 48);
        na = wmma16(a0, bn0, na);
        da = wmma16(a0, bd0, da);
        na = wmma16(a1, bn1, na);
        da = wmma16(a1, bd1, da);
#pragma unroll
        for (int r = 0; r < 8; ++r)
            qa[jt][r] = qa[jt][r] * na[r] / da[r];
    }

    // mid -> staging (reuse ek region; wave-private, DS ops are in order)
#pragma unroll
    for (int jt = 0; jt < 4; ++jt) {
#pragma unroll
        for (int r = 0; r < 8; r += 2) {
            h2 pk;
            pk.x = (_Float16)qa[jt][r];
            pk.y = (_Float16)qa[jt][r + 1];
            *reinterpret_cast<h2*>(ekp + sbase + 16 * jt + 8 * hi8 + r) = pk;
        }
    }

    // ---- stage 3 B fragments (mid): shared by all o-tiles ----
    v16h bm0 = ld_frag16(ekp, bo0,      bo0 + 16);
    v16h bm1 = ld_frag16(ekp, bo0 + 32, bo0 + 48);

    // ---- stage 3: out = Wm@mid ; store f32 ----
#pragma unroll
    for (int ot = 0; ot < 8; ++ot) {
        v8f acc = {0.f, 0.f, 0.f, 0.f, 0.f, 0.f, 0.f, 0.f};
        const int ao0 = (16 * ot + tl) * ICH + 8 * hi8;
        v16h a0 = ld_frag16(wsh + WM_OFF, ao0,      ao0 + 16);
        v16h a1 = ld_frag16(wsh + WM_OFF, ao0 + 32, ao0 + 48);
        acc = wmma16(a0, bm0, acc);
        acc = wmma16(a1, bm1, acc);
#pragma unroll
        for (int r = 0; r < 8; ++r) {
            const int orow = 16 * ot + r + 8 * hi8;
            outg[(size_t)(b * CCH + orow) * TT + t0 + w * 16 + tl] = acc[r];
        }
    }
}

// ---------------------------------------------------------------------------
extern "C" void kernel_launch(void* const* d_in, const int* in_sizes, int n_in,
                              void* d_out, int out_size, void* d_ws, size_t ws_size,
                              hipStream_t stream) {
    const float* x  = (const float*)d_in[0];
    const float* Wq = (const float*)d_in[1];
    const float* Wk = (const float*)d_in[2];
    const float* Wv = (const float*)d_in[3];
    const float* Bb = (const float*)d_in[4];
    const float* Wm = (const float*)d_in[5];
    float*      out = (float*)d_out;
    _Float16*   wsh = (_Float16*)d_ws;

    aft_prep<<<(WS_TOT + 255) / 256, 256, 0, stream>>>(Wq, Wk, Wv, Bb, Wm, wsh);
    aft_main<<<dim3(TT / T_TILE, BSZ), 128, 0, stream>>>(x, wsh, out);
}
